// BinaryConv2D_44908178047612
// MI455X (gfx1250) — 8-bit matrix operand order repaired and checked by reading, compiled, not run on hardware
//
#include <hip/hip_runtime.h>

typedef __attribute__((ext_vector_type(8))) int v8i;

#define NIMG   64
#define HH     56
#define WW     56
#define CIN    128
#define COUT   256
#define MTOT   (NIMG * HH * WW)
#define KSTEPS 18
#define NTILES (COUT / 16)

#define XQ_BYTES ((size_t)NIMG * HH * WW * CIN)
#define WQ_SLOTS (KSTEPS * NTILES * 32)
#define WQ_BYTES ((size_t)WQ_SLOTS * 32)

__global__ void binarize_x_kernel(const float* __restrict__ x,
                                  signed char* __restrict__ xq, int n4) {
    int i = blockIdx.x * blockDim.x + threadIdx.x;
    if (i >= n4) return;
    float4 v = ((const float4*)x)[i];
    int b0 = (v.x >= 0.f) ? 1 : 0xFF;
    int b1 = (v.y >= 0.f) ? 1 : 0xFF;
    int b2 = (v.z >= 0.f) ? 1 : 0xFF;
    int b3 = (v.w >= 0.f) ? 1 : 0xFF;
    ((int*)xq)[i] = (b0 & 0xFF) | ((b1 & 0xFF) << 8) |
                    ((b2 & 0xFF) << 16) | ((b3 & 0xFF) << 24);
}

__global__ void repack_w_kernel(const float* __restrict__ w,
                                signed char* __restrict__ wq) {
    int t = blockIdx.x * blockDim.x + threadIdx.x;
    if (t >= WQ_SLOTS) return;
    int lane  = t & 31;
    int ntile = (t >> 5) & 15;
    int kstep = t >> 9;
    int n  = ntile * 16 + (lane & 15);
    int kb = (lane >= 16) ? 16 : 0;
    signed char* dst = wq + (size_t)t * 32;
    #pragma unroll
    for (int vv = 0; vv < 8; ++vv) {
        #pragma unroll
        for (int b = 0; b < 4; ++b) {
            int klocal = (vv < 4) ? (kb + vv * 4 + b)
                                  : (kb + 32 + (vv - 4) * 4 + b);
            int k   = kstep * 64 + klocal;
            int tap = k >> 7;
            int cin = k & 127;
            float wv = w[((size_t)(tap * CIN + cin)) * COUT + n];
            dst[vv * 4 + b] = (wv >= 0.f) ? (signed char)1 : (signed char)-1;
        }
    }
}

__global__ void __launch_bounds__(512)
conv_wmma_kernel(const signed char* __restrict__ xq,
                 const signed char* __restrict__ wq,
                 float* __restrict__ out) {
    const int lane  = threadIdx.x & 31;
    const int ntile = threadIdx.x >> 5;
    const int mtile = blockIdx.x;

    const int m    = mtile * 16 + (lane & 15);
    const int nimg = m / (HH * WW);
    const int rem  = m - nimg * (HH * WW);
    const int oh   = rem / WW;
    const int ow   = rem - oh * WW;

    const int kb8 = (lane >= 16) ? 16 : 0;

    const signed char* bslot = wq + ((size_t)(ntile * 32 + lane)) * 32;
    const size_t bstride = (size_t)NTILES * 32 * 32;

    v8i acc = {};

    #pragma unroll
    for (int kstep = 0; kstep < KSTEPS; ++kstep) {
        v8i bf = *(const v8i*)(bslot + (size_t)kstep * bstride);

        const int tap = kstep >> 1;
        const int kh  = tap / 3;
        const int kw  = tap - kh * 3;
        const int ih  = oh + kh - 1;
        const int iw  = ow + kw - 1;
        const bool inb = ((unsigned)ih < (unsigned)HH) &&
                         ((unsigned)iw < (unsigned)WW);
        const int ihc = inb ? ih : 0;
        const int iwc = inb ? iw : 0;
        const int cbase = (kstep & 1) << 6;
        const signed char* ap =
            xq + ((size_t)((nimg * HH + ihc) * WW + iwc)) * CIN + cbase + kb8;

        v8i af;
        #pragma unroll
        for (int c = 0; c < 4; ++c) {
            long long vv = 0;
            if (inb) vv = ((const long long*)ap)[(c & 1) + (c >> 1) * 4];
            af[2 * c]     = (int)vv;
            af[2 * c + 1] = (int)(vv >> 32);
        }

        acc = __builtin_amdgcn_wmma_i32_16x16x64_iu8(
            true, af, true, bf, acc, false, false);
    }

    const int mhalf = (lane >> 4) * 8;
    float* obase = out + (size_t)(mtile * 16) * COUT + ntile * 16 + (lane & 15);
    #pragma unroll
    for (int r = 0; r < 8; ++r) {
        obase[(size_t)(mhalf + r) * COUT] = (float)acc[r];
    }
}

extern "C" void kernel_launch(void* const* d_in, const int* in_sizes, int n_in,
                              void* d_out, int out_size, void* d_ws, size_t ws_size,
                              hipStream_t stream) {
    const float* x = (const float*)d_in[0];
    const float* w = (const float*)d_in[1];
    float* out = (float*)d_out;

    signed char* xq = (signed char*)d_ws;
    signed char* wq = (signed char*)d_ws + XQ_BYTES;

    {
        int n4 = (int)(XQ_BYTES / 4);
        int blocks = (n4 + 255) / 256;
        binarize_x_kernel<<<blocks, 256, 0, stream>>>(x, xq, n4);
    }
    {
        int blocks = (WQ_SLOTS + 255) / 256;
        repack_w_kernel<<<blocks, 256, 0, stream>>>(w, wq);
    }
    {
        int blocks = MTOT / 16;
        conv_wmma_kernel<<<blocks, 512, 0, stream>>>(xq, wq, out);
    }
}
